// GCNFeatureUpdater_39848706572466
// MI455X (gfx1250) — matrix operand order repaired and checked by reading, not compiled, not run on hardware
//
#include <hip/hip_runtime.h>
#include <hip/hip_bf16.h>

// ---- types ----------------------------------------------------------------
typedef __bf16 bf16;
typedef __attribute__((ext_vector_type(16))) bf16  v16bf;
typedef __attribute__((ext_vector_type(8)))  bf16  v8bf;
typedef __attribute__((ext_vector_type(8)))  float v8f;
typedef long long eidx_t;   // reference: edge_index dtype = int64

#define THREADS 256

static inline int cdiv_i(long long a, long long b) { return (int)((a + b - 1) / b); }

// ---- degree / normalization ----------------------------------------------
__global__ void k_init_deg(float* __restrict__ deg, int n) {
  int i = blockIdx.x * blockDim.x + threadIdx.x;
  if (i < n) deg[i] = 1.0f;                      // self-loop contributes 1
}

__global__ void k_deg(const eidx_t* __restrict__ ei, float* __restrict__ deg, int E) {
  int e = blockIdx.x * blockDim.x + threadIdx.x;
  if (e < E) unsafeAtomicAdd(&deg[(int)ei[(size_t)E + e]], 1.0f);  // dst row
}

__global__ void k_dinv(const float* __restrict__ deg, float* __restrict__ dinv, int n) {
  int i = blockIdx.x * blockDim.x + threadIdx.x;
  if (i < n) dinv[i] = rsqrtf(deg[i]);
}

__global__ void k_norm(const eidx_t* __restrict__ ei, const float* __restrict__ dinv,
                       float* __restrict__ norm, int E) {
  int e = blockIdx.x * blockDim.x + threadIdx.x;
  if (e < E) {
    int s = (int)ei[e];
    int d = (int)ei[(size_t)E + e];
    norm[e] = dinv[s] * dinv[d];
  }
}

// ---- precision conversion -------------------------------------------------
__global__ void k_cvt4(const float* __restrict__ x, bf16* __restrict__ xb, int n4) {
  int i = blockIdx.x * blockDim.x + threadIdx.x;
  if (i < n4) {
    float4 v = ((const float4*)x)[i];
    bf16* o = xb + 4 * (size_t)i;
    o[0] = (bf16)v.x; o[1] = (bf16)v.y; o[2] = (bf16)v.z; o[3] = (bf16)v.w;
  }
}

// wt[n*K + k] = (bf16) W[k*Ncol + n]   (transpose so B columns are contiguous)
__global__ void k_cvt_wT(const float* __restrict__ W, bf16* __restrict__ wt, int K, int Ncol) {
  int i = blockIdx.x * blockDim.x + threadIdx.x;
  if (i < K * Ncol) {
    int k = i / Ncol, n = i % Ncol;
    wt[(size_t)n * K + k] = (bf16)W[i];
  }
}

// ---- WMMA GEMM: C[M,Ncol] = A[M,K](bf16) x Bt[Ncol,K](bf16, transposed) ----
// One wave computes one 16x16 tile of C via v_wmma_f32_16x16x32_bf16 over K/32 steps.
__global__ void k_gemm_bf16(const bf16* __restrict__ A, const bf16* __restrict__ Bt,
                            float* __restrict__ C, int M, int K, int Ncol) {
  int wid  = (blockIdx.x * blockDim.x + threadIdx.x) >> 5;
  int lane = threadIdx.x & 31;
  int ntiles = Ncol >> 4;
  int total  = (M >> 4) * ntiles;
  if (wid >= total) return;                      // whole-wave uniform: EXEC stays all-1s

  int m0 = (wid / ntiles) << 4;
  int n0 = (wid % ntiles) << 4;
  int half = lane >> 4;                          // 0: lanes 0-15, 1: lanes 16-31
  int row  = m0 + (lane & 15);                   // A fragment: row per lane
  int col  = n0 + (lane & 15);                   // B fragment: column per lane
  int akoff = half * 16;                         // A: K 0..15 / 16..31 (+8 for hi 8 elems)
  int bkoff = half * 16;                         // B: K 0..15 / 16..31

  v8f c = {};
  for (int k0 = 0; k0 < K; k0 += 32) {
    // A 16x32 bf16 fragment (contiguous: lane half h holds K 16h..16h+15)
    v8bf alo = *(const v8bf*)(A + (size_t)row * K + k0 + akoff);
    v8bf ahi = *(const v8bf*)(A + (size_t)row * K + k0 + 8 + akoff);
    v16bf a;
#pragma unroll
    for (int j = 0; j < 8; ++j) { a[j] = alo[j]; a[j + 8] = ahi[j]; }
    // B 32x16 bf16 fragment: contiguous 16 elems of the (transposed) column
    v16bf b = *(const v16bf*)(Bt + (size_t)col * K + k0 + bkoff);
    c = __builtin_amdgcn_wmma_f32_16x16x32_bf16(false, a, false, b, (short)0, c,
                                                false, false);
  }
  // D layout: lanes 0-15 -> rows m0+0..7, lanes 16-31 -> rows m0+8..15, col = n0+(lane&15)
#pragma unroll
  for (int j = 0; j < 8; ++j)
    C[(size_t)(m0 + half * 8 + j) * Ncol + col] = c[j];
}

// ---- aggregation ----------------------------------------------------------
// out[i,f] = dinv[i]^2 * t[i,f] + bias[f]   (self-loop message + bias, full overwrite)
__global__ void k_selfagg(const float* __restrict__ t, const float* __restrict__ dinv,
                          const float* __restrict__ bias, float* __restrict__ out,
                          int n, int F) {
  int i = blockIdx.x * blockDim.x + threadIdx.x;
  if (i < n * F) {
    int node = i / F, f = i - node * F;
    float d = dinv[node];
    out[i] = d * d * t[i] + bias[f];
  }
}

// out[dst] += norm[e] * t[src]; F/4 lanes' worth of float4 per edge, TPE threads/edge
__global__ void k_edge_agg(const eidx_t* __restrict__ ei, const float* __restrict__ norm,
                           const float* __restrict__ t, float* __restrict__ out,
                           int E, int F, int TPE) {
  int tid = blockIdx.x * blockDim.x + threadIdx.x;
  int e = tid / TPE;
  if (e >= E) return;
  int f = (tid - e * TPE) * 4;
  __builtin_prefetch(&ei[e + 8192], 0, 1);       // global_prefetch on the edge stream
  int s = (int)ei[e];
  int d = (int)ei[(size_t)E + e];
  float w = norm[e];
  float4 v = *(const float4*)(t + (size_t)s * F + f);
  float* o = out + (size_t)d * F + f;
  unsafeAtomicAdd(o + 0, w * v.x);
  unsafeAtomicAdd(o + 1, w * v.y);
  unsafeAtomicAdd(o + 2, w * v.z);
  unsafeAtomicAdd(o + 3, w * v.w);
}

__global__ void k_relu_bf16(const float* __restrict__ a, bf16* __restrict__ hb, int n) {
  int i = blockIdx.x * blockDim.x + threadIdx.x;
  if (i < n) {
    float v = a[i];
    hb[i] = (bf16)(v > 0.0f ? v : 0.0f);
  }
}

// ---- launch ---------------------------------------------------------------
extern "C" void kernel_launch(void* const* d_in, const int* in_sizes, int n_in,
                              void* d_out, int out_size, void* d_ws, size_t ws_size,
                              hipStream_t stream) {
  const float*  x  = (const float*)d_in[0];
  const eidx_t* ei = (const eidx_t*)d_in[1];
  const float*  W1 = (const float*)d_in[2];
  const float*  b1 = (const float*)d_in[3];
  const float*  W2 = (const float*)d_in[4];
  const float*  b2 = (const float*)d_in[5];
  float* out = (float*)d_out;

  const int NFEAT = 128, NHID = 64;
  const int N = in_sizes[0] / NFEAT;       // 100000
  const int E = in_sizes[1] / 2;           // 1600000

  // workspace carve-out (256B aligned slabs)
  size_t off = 0;
  char* ws = (char*)d_ws;
  auto take = [&](size_t bytes) -> void* {
    void* p = ws + off;
    off += (bytes + 255) & ~(size_t)255;
    return p;
  };
  float* deg  = (float*)take((size_t)N * 4);
  float* dinv = (float*)take((size_t)N * 4);
  float* norm = (float*)take((size_t)E * 4);
  bf16*  xb   = (bf16*) take((size_t)N * NFEAT * 2);
  bf16*  w1t  = (bf16*) take((size_t)NFEAT * NHID * 2);
  bf16*  w2t  = (bf16*) take((size_t)NHID * NFEAT * 2);
  float* t1   = (float*)take((size_t)N * NHID * 4);   // 25.6MB (multiple of 256)
  float* agg1 = (float*)take((size_t)N * NHID * 4);   // 25.6MB, contiguous after t1
  bf16*  h1b  = (bf16*) take((size_t)N * NHID * 2);
  float* t2   = t1;  // reuse t1|agg1 region (both dead before GEMM2) -> 51.2MB
  (void)ws_size; (void)n_in; (void)out_size;

  // 1) symmetric normalization
  k_init_deg<<<cdiv_i(N, THREADS), THREADS, 0, stream>>>(deg, N);
  k_deg     <<<cdiv_i(E, THREADS), THREADS, 0, stream>>>(ei, deg, E);
  k_dinv    <<<cdiv_i(N, THREADS), THREADS, 0, stream>>>(deg, dinv, N);
  k_norm    <<<cdiv_i(E, THREADS), THREADS, 0, stream>>>(ei, dinv, norm, E);

  // 2) bf16 operands (weights pre-transposed for contiguous B fragments)
  k_cvt4  <<<cdiv_i((size_t)N * NFEAT / 4, THREADS), THREADS, 0, stream>>>(x, xb, N * NFEAT / 4);
  k_cvt_wT<<<cdiv_i(NFEAT * NHID, THREADS), THREADS, 0, stream>>>(W1, w1t, NFEAT, NHID);
  k_cvt_wT<<<cdiv_i(NHID * NFEAT, THREADS), THREADS, 0, stream>>>(W2, w2t, NHID, NFEAT);

  // 3) layer 1: t1 = x @ W1 (WMMA bf16 -> f32)
  {
    long long tiles = ((long long)N / 16) * (NHID / 16);     // 25000 (multiple of 8)
    k_gemm_bf16<<<cdiv_i(tiles * 32, THREADS), THREADS, 0, stream>>>(xb, w1t, t1, N, NFEAT, NHID);
  }
  // 4) aggregate: agg1 = D^-1/2 A D^-1/2 t1 + b1
  k_selfagg <<<cdiv_i((size_t)N * NHID, THREADS), THREADS, 0, stream>>>(t1, dinv, b1, agg1, N, NHID);
  k_edge_agg<<<cdiv_i((size_t)E * (NHID / 4), THREADS), THREADS, 0, stream>>>(ei, norm, t1, agg1, E, NHID, NHID / 4);
  // 5) relu + bf16
  k_relu_bf16<<<cdiv_i((size_t)N * NHID, THREADS), THREADS, 0, stream>>>(agg1, h1b, N * NHID);

  // 6) layer 2: t2 = h1 @ W2 (WMMA bf16 -> f32)
  {
    long long tiles = ((long long)N / 16) * (NFEAT / 16);    // 50000 (multiple of 8)
    k_gemm_bf16<<<cdiv_i(tiles * 32, THREADS), THREADS, 0, stream>>>(h1b, w2t, t2, N, NHID, NFEAT);
  }
  // 7) aggregate into output: out = D^-1/2 A D^-1/2 t2 + b2
  k_selfagg <<<cdiv_i((size_t)N * NFEAT, THREADS), THREADS, 0, stream>>>(t2, dinv, b2, out, N, NFEAT);
  k_edge_agg<<<cdiv_i((size_t)E * (NFEAT / 4), THREADS), THREADS, 0, stream>>>(ei, norm, t2, out, E, NFEAT, NFEAT / 4);
}
